// AttentionLayer_36532991820656
// MI455X (gfx1250) — compile-verified, matrix operand order checked by reading, not run on hardware
//
#include <hip/hip_runtime.h>
#include <hip/hip_bf16.h>

typedef __attribute__((ext_vector_type(16))) __bf16 v16bf;
typedef __attribute__((ext_vector_type(8)))  float  v8f;

#define B_DIM 32
#define S_DIM 2048
#define D_DIM 1024
#define A_DIM 1024

#define TM 64            // seq rows per workgroup (4 x 16 M-subtiles)
#define NPASS 512        // N columns per pass (2 passes cover A=1024)
#define KSTEP 32         // bf16 WMMA K
#define LDSA_STRIDE 40   // bf16 elems per row (80B, 16B aligned, bank-skewed)
#define LDSB_STRIDE 40

__device__ __forceinline__ float fast_tanhf(float x) {
    // tanh(x) = 1 - 2/(exp(2x)+1); branchless, saturates correctly at +/-1
    float e = __expf(2.0f * x);
    return 1.0f - __fdividef(2.0f, e + 1.0f);
}

// ---------- Kernel 0a: Wk f32 [D][A] -> WkT bf16 [A][D] (transpose+convert) ----
__global__ __launch_bounds__(256)
void wk_transpose_bf16(const float* __restrict__ Wk, __bf16* __restrict__ WkT) {
    __shared__ float tile[32][33];
    const int a0 = blockIdx.x * 32;
    const int d0 = blockIdx.y * 32;
    const int tx = threadIdx.x;   // 0..31
    const int ty = threadIdx.y;   // 0..7
#pragma unroll
    for (int i = 0; i < 4; ++i) {
        int dl = ty + i * 8;
        tile[dl][tx] = Wk[(size_t)(d0 + dl) * A_DIM + a0 + tx];
    }
    __syncthreads();
#pragma unroll
    for (int i = 0; i < 4; ++i) {
        int al = ty + i * 8;
        WkT[(size_t)(a0 + al) * D_DIM + d0 + tx] = (__bf16)tile[tx][al];
    }
}

// ---------- Kernel 0b: qproj[b][a] = query[b]·Wq[:,a] + bq[a] ------------------
__global__ __launch_bounds__(256)
void qproj_kernel(const float* __restrict__ q, const float* __restrict__ Wq,
                  const float* __restrict__ bq, float* __restrict__ qp) {
    const int a = blockIdx.x * 256 + threadIdx.x;
    const int b = blockIdx.y;
    const float* qrow = q + (size_t)b * D_DIM;
    float acc = 0.f;
    for (int d = 0; d < D_DIM; ++d)
        acc += qrow[d] * Wq[(size_t)d * A_DIM + a];
    qp[(size_t)b * A_DIM + a] = acc + bq[a];
}

// ---------- Kernel 1: fused key@Wk + bias + qproj -> tanh -> ·Wa  (WMMA) -------
// 8 waves; each wave: 4(M) x 4(N) = 16 accumulator tiles (64 rows x 64 cols),
// => 16 LDS fragment loads per 16 WMMAs (1.0 b128/WMMA, LDS-bandwidth balanced).
__global__ __launch_bounds__(256)
void scores_kernel(const float* __restrict__ key, const __bf16* __restrict__ WkT,
                   const float* __restrict__ qproj, const float* __restrict__ bk,
                   const float* __restrict__ Wa, float* __restrict__ scores) {
    __shared__ __attribute__((aligned(16))) __bf16 Alds[TM * LDSA_STRIDE];     // 5 KB
    __shared__ __attribute__((aligned(16))) __bf16 Blds[NPASS * LDSB_STRIDE];  // 40 KB
    __shared__ float scoreLDS[TM];

    const int tid  = threadIdx.x;
    const int b    = blockIdx.y;
    const int s0   = blockIdx.x * TM;
    const int wave = tid >> 5;
    const int lane = tid & 31;
    const int lrow = lane & 15;
    const int half = lane >> 4;      // 0: K lo-half / M 0..7 ; 1: K hi-half / M 8..15
    const int ncol0 = wave * 64;     // this wave's 64-column strip within the pass

    if (tid < TM) scoreLDS[tid] = 0.f;
    __syncthreads();

    const float* keyBase = key + ((size_t)b * S_DIM + s0) * D_DIM;

    for (int npass = 0; npass < A_DIM; npass += NPASS) {
        v8f acc[4][4] = {};   // [mi][ni]

        for (int k0 = 0; k0 < D_DIM; k0 += KSTEP) {
            __syncthreads();   // previous iteration's LDS reads complete
            // --- stage A: key[s0..s0+63][k0..k0+31] fp32 -> bf16 ---
            {
                int r = tid >> 2;          // 0..63
                int c = (tid & 3) * 8;     // 0,8,16,24
                const float4* src = (const float4*)(keyBase + (size_t)r * D_DIM + k0 + c);
                float4 f0 = src[0];
                float4 f1 = src[1];
                __bf16* dst = &Alds[r * LDSA_STRIDE + c];
                dst[0] = (__bf16)f0.x; dst[1] = (__bf16)f0.y;
                dst[2] = (__bf16)f0.z; dst[3] = (__bf16)f0.w;
                dst[4] = (__bf16)f1.x; dst[5] = (__bf16)f1.y;
                dst[6] = (__bf16)f1.z; dst[7] = (__bf16)f1.w;
            }
            // --- stage B: WkT[npass..npass+511][k0..k0+31] (already bf16) ---
#pragma unroll
            for (int rr = 0; rr < 2; ++rr) {
                int n = tid + rr * 256;    // 0..511
                const uint4* src = (const uint4*)(WkT + (size_t)(npass + n) * D_DIM + k0);
                uint4* dst = (uint4*)&Blds[n * LDSB_STRIDE];
                dst[0] = src[0]; dst[1] = src[1]; dst[2] = src[2]; dst[3] = src[3];
            }
            __syncthreads();

            union FragB16 { v16bf v; uint4 q[2]; } fa[4], fb;
            // A fragments: rows M = mi*16 + lrow; K = {half*8+0..7, 16+half*8+0..7}
#pragma unroll
            for (int mi = 0; mi < 4; ++mi) {
                const uint4* ap =
                    (const uint4*)&Alds[(mi * 16 + lrow) * LDSA_STRIDE + half * 8];
                fa[mi].q[0] = ap[0];
                fa[mi].q[1] = ap[2];   // +16 bf16 elems = +32B
            }
#pragma unroll
            for (int ni = 0; ni < 4; ++ni) {
                const uint4* bp =
                    (const uint4*)&Blds[(ncol0 + ni * 16 + lrow) * LDSB_STRIDE + half * 8];
                fb.q[0] = bp[0];
                fb.q[1] = bp[2];
#pragma unroll
                for (int mi = 0; mi < 4; ++mi) {
                    acc[mi][ni] = __builtin_amdgcn_wmma_f32_16x16x32_bf16(
                        false, fa[mi].v, false, fb.v, (short)0, acc[mi][ni], false, false);
                }
            }
        }

        // --- epilogue: tanh(key_proj + qproj + bk) · Wa, reduce over N ---
        float partial[4][8];
#pragma unroll
        for (int mi = 0; mi < 4; ++mi)
#pragma unroll
            for (int r = 0; r < 8; ++r) partial[mi][r] = 0.f;

#pragma unroll
        for (int ni = 0; ni < 4; ++ni) {
            int n = npass + ncol0 + ni * 16 + lrow;      // global A-column
            float base = qproj[(size_t)b * A_DIM + n] + bk[n];
            float wav  = Wa[n];
#pragma unroll
            for (int mi = 0; mi < 4; ++mi) {
#pragma unroll
                for (int r = 0; r < 8; ++r) {
                    float v = fast_tanhf(acc[mi][ni][r] + base);
                    partial[mi][r] += v * wav;
                }
            }
        }
#pragma unroll
        for (int mi = 0; mi < 4; ++mi) {
#pragma unroll
            for (int r = 0; r < 8; ++r) {
                float p = partial[mi][r];
                p += __shfl_xor(p, 1);
                p += __shfl_xor(p, 2);
                p += __shfl_xor(p, 4);
                p += __shfl_xor(p, 8);   // stays within each 16-lane half
                if (lrow == 0) atomicAdd(&scoreLDS[mi * 16 + half * 8 + r], p);
            }
        }
    }

    __syncthreads();
    if (tid < TM) scores[(size_t)b * S_DIM + s0 + tid] = scoreLDS[tid];
}

// ---------- Kernel 2a: softmax over S, write att_weights -----------------------
__global__ __launch_bounds__(256)
void softmax_kernel(const float* __restrict__ scores, const float* __restrict__ ba,
                    float* __restrict__ att) {
    __shared__ float red[256];
    const int b = blockIdx.x;
    const int tid = threadIdx.x;
    const float bav = ba[0];
    float local[8];
    float m = -INFINITY;
#pragma unroll
    for (int i = 0; i < 8; ++i) {
        float v = scores[(size_t)b * S_DIM + tid + i * 256] + bav;
        local[i] = v;
        m = fmaxf(m, v);
    }
    red[tid] = m; __syncthreads();
    for (int off = 128; off > 0; off >>= 1) {
        if (tid < off) red[tid] = fmaxf(red[tid], red[tid + off]);
        __syncthreads();
    }
    float gmax = red[0]; __syncthreads();
    float sum = 0.f;
#pragma unroll
    for (int i = 0; i < 8; ++i) { local[i] = __expf(local[i] - gmax); sum += local[i]; }
    red[tid] = sum; __syncthreads();
    for (int off = 128; off > 0; off >>= 1) {
        if (tid < off) red[tid] += red[tid + off];
        __syncthreads();
    }
    float inv = 1.f / red[0];
#pragma unroll
    for (int i = 0; i < 8; ++i)
        att[(size_t)b * S_DIM + tid + i * 256] = local[i] * inv;
}

// ---------- Kernel 2b: context[b][d] = sum_s att[b][s] * value[b][s][d] --------
__global__ __launch_bounds__(128)
void context_kernel(const float* __restrict__ value, const float* __restrict__ att,
                    float* __restrict__ ctx) {
    __shared__ float w[S_DIM];   // 8 KB
    const int b = blockIdx.y;
    const int d = blockIdx.x * 128 + threadIdx.x;
    for (int i = threadIdx.x; i < S_DIM; i += 128)
        w[i] = att[(size_t)b * S_DIM + i];
    __syncthreads();
    const float* vb = value + ((size_t)b * S_DIM) * D_DIM + d;
    float acc = 0.f;
#pragma unroll 4
    for (int s = 0; s < S_DIM; ++s)
        acc += w[s] * vb[(size_t)s * D_DIM];
    ctx[(size_t)b * D_DIM + d] = acc;
}

// -------------------------------------------------------------------------------
extern "C" void kernel_launch(void* const* d_in, const int* in_sizes, int n_in,
                              void* d_out, int out_size, void* d_ws, size_t ws_size,
                              hipStream_t stream) {
    (void)in_sizes; (void)n_in; (void)out_size; (void)ws_size;
    const float* query = (const float*)d_in[0];   // [B, D]
    const float* key   = (const float*)d_in[1];   // [B, S, D]
    const float* value = (const float*)d_in[2];   // [B, S, D]
    const float* Wq    = (const float*)d_in[3];   // [D, A]
    const float* bq    = (const float*)d_in[4];   // [A]
    const float* Wk    = (const float*)d_in[5];   // [D, A]
    const float* bk    = (const float*)d_in[6];   // [A]
    const float* Wa    = (const float*)d_in[7];   // [A, 1]
    const float* ba    = (const float*)d_in[8];   // [1]

    float* out = (float*)d_out;
    float* ctx = out;                          // [B, 1, D]  -> 32768 floats
    float* att = out + (size_t)B_DIM * D_DIM;  // [B, S, 1]  -> 65536 floats

    char* ws = (char*)d_ws;
    __bf16* WkT   = (__bf16*)ws;                                  // 2 MB
    float*  qp    = (float*)(ws + (size_t)A_DIM * D_DIM * 2);     // 128 KB
    float*  score = (float*)(ws + (size_t)A_DIM * D_DIM * 2
                                + (size_t)B_DIM * A_DIM * 4);     // 256 KB

    wk_transpose_bf16<<<dim3(A_DIM / 32, D_DIM / 32), dim3(32, 8), 0, stream>>>(Wk, WkT);
    qproj_kernel<<<dim3(A_DIM / 256, B_DIM), 256, 0, stream>>>(query, Wq, bq, qp);
    scores_kernel<<<dim3(S_DIM / TM, B_DIM), 256, 0, stream>>>(key, WkT, qp, bk, Wa, score);
    softmax_kernel<<<B_DIM, 256, 0, stream>>>(score, ba, att);
    context_kernel<<<dim3(D_DIM / 128, B_DIM), 128, 0, stream>>>(value, att, ctx);
}
